// ExpressionPerformer_66872640798733
// MI455X (gfx1250) — matrix operand order repaired and checked by reading, not compiled, not run on hardware
//
#include <hip/hip_runtime.h>
#include <hip/hip_bf16.h>
#include <math.h>

// ---------------------------------------------------------------------------
// ExpressionPerformer for MI455X (gfx1250, wave32, WMMA).
// All GEMMs + the chunked causal linear-attention scan run on
// v_wmma_f32_16x16x32_bf16 (f32 accumulate).  GEMM A tiles are staged into
// LDS with global_load_async_to_lds_b128 (ASYNCcnt); B (weight) tiles are
// DMA'd by the Tensor Data Mover (tensor_load_to_lds, TENSORcnt) with
// hardware LDS row padding.  B=4,G=2048,D=256,H=8.
// ---------------------------------------------------------------------------

typedef __attribute__((ext_vector_type(16))) __bf16 v16bf;
typedef __attribute__((ext_vector_type(8)))  float  v8f;
typedef __attribute__((ext_vector_type(4)))  int    v4i;
typedef __attribute__((ext_vector_type(8)))  int    v8i;

__device__ __forceinline__ __bf16 usbf(unsigned short u) {
  return __builtin_bit_cast(__bf16, u);
}
__device__ __forceinline__ unsigned short f2bfu(float f) {  // RNE f32 -> bf16 bits
  unsigned int u = __builtin_bit_cast(unsigned int, f);
  u += 0x7fffu + ((u >> 16) & 1u);
  return (unsigned short)(u >> 16);
}
__device__ __forceinline__ float bf2f(unsigned short s) {
  unsigned int u = ((unsigned int)s) << 16;
  return __builtin_bit_cast(float, u);
}
__device__ __forceinline__ void put2(v16bf& d, int i, unsigned int p) {
  d[i]     = usbf((unsigned short)(p & 0xffffu));
  d[i + 1] = usbf((unsigned short)(p >> 16));
}
__device__ __forceinline__ v8f wmma_bf16(v16bf a, v16bf b, v8f c) {
  // (neg_a, A, neg_b, B, c_mod, C, reuse_a, reuse_b)
  return __builtin_amdgcn_wmma_f32_16x16x32_bf16(false, a, false, b,
                                                 (short)0, c, false, false);
}
// Async global->LDS 16-byte copy (per active lane); tracked by ASYNCcnt.
__device__ __forceinline__ void async_g2l_b128(const void* lds, const void* gptr) {
  const unsigned lds_off = (unsigned)(size_t)lds;      // addr[31:0] == LDS offset
  const unsigned long long ga = (unsigned long long)(size_t)gptr;
  asm volatile("global_load_async_to_lds_b128 %0, %1, off"
               :: "v"(lds_off), "v"(ga) : "memory");
}
__device__ __forceinline__ void wait_async0() {
  asm volatile("s_wait_asynccnt 0x0" ::: "memory");
}
// Tensor Data Mover: DMA a 64-row x 32-col bf16 tile (row stride = strideElems)
// into LDS with 32 B of padding after every 64 B row (-> 96 B LDS rows).
// 2-D descriptor: D# groups 0 (4 SGPRs) + 1 (8 SGPRs); VADDR2/3 = NULL.
__device__ __forceinline__ void tdm_load_64x32_bf16(
    const void* lds, const unsigned short* g, int strideElems) {
  const unsigned long long ga = (unsigned long long)(size_t)g;
  v4i g0;
  g0[0] = 1;                                            // count=1, user D#
  g0[1] = (int)(unsigned)(size_t)lds;                   // lds_addr
  g0[2] = (int)(unsigned)(ga & 0xffffffffull);          // global_addr[31:0]
  g0[3] = (int)(((ga >> 32) & 0x01ffffffull) | (2u << 30));  // addr[56:32]|type=2
  v8i g1;
  g1[0] = (1 << 16) | (1 << 20) | (3 << 22) | (7 << 25);
  //       data_size=2B | pad_en | pad_interval=16DW | pad_amount=8DW
  g1[1] = (32 << 16);          // tensor_dim0[15:0] = 32   (bits 79:48)
  g1[2] = (64 << 16);          // dim0 hi = 0, tensor_dim1[15:0] = 64
  g1[3] = (32 << 16);          // dim1 hi = 0, tile_dim0 = 32
  g1[4] = 64;                  // tile_dim1 = 64, tile_dim2 = 0
  g1[5] = strideElems;         // tensor_dim0_stride[31:0] (data_size units)
  g1[6] = 0;                   // stride hi, tensor_dim1_stride lo
  g1[7] = 0;
  asm volatile("tensor_load_to_lds %0, %1" :: "s"(g0), "s"(g1) : "memory");
}

// ---------------------------------------------------------------------------
// Rotary expression embedding: h[token][d] = gene_emb[g][d] + ree
// ---------------------------------------------------------------------------
__global__ __launch_bounds__(256) void embed_kernel(
    const float* __restrict__ x, const float* __restrict__ ge,
    float* __restrict__ h) {
  const int token = blockIdx.x;         // 0..8191
  const int d     = threadIdx.x;        // 0..255
  const int g     = token & 2047;
  const float xv  = x[token];
  float ree = 0.0f;
  if (xv != -10.0f) {
    const int   i    = d & 127;
    const float invf = expf(-(float)(2 * i) * (1.0f / 256.0f) * logf(100.0f));
    const float f    = xv * invf;
    ree = (d < 128) ? sinf(f) : cosf(f);
  }
  h[(size_t)token * 256 + d] = ge[(size_t)g * 256 + d] + ree;
}

// ---------------------------------------------------------------------------
// LayerNorm: one wave per token, 8 elems/lane, shuffle reductions -> bf16 z
// ---------------------------------------------------------------------------
__global__ __launch_bounds__(256) void ln_kernel(
    const float* __restrict__ h, const float* __restrict__ gam,
    const float* __restrict__ bet, unsigned short* __restrict__ z, int T) {
  const int wid  = (blockIdx.x * blockDim.x + threadIdx.x) >> 5;
  const int lane = threadIdx.x & 31;
  if (wid >= T) return;
  const float* row = h + (size_t)wid * 256;
  float v[8];
  float s = 0.0f;
#pragma unroll
  for (int i = 0; i < 8; ++i) { v[i] = row[i * 32 + lane]; s += v[i]; }
#pragma unroll
  for (int m = 16; m >= 1; m >>= 1) s += __shfl_xor(s, m);
  const float mu = s * (1.0f / 256.0f);
  float var = 0.0f;
#pragma unroll
  for (int i = 0; i < 8; ++i) { const float d = v[i] - mu; var += d * d; }
#pragma unroll
  for (int m = 16; m >= 1; m >>= 1) var += __shfl_xor(var, m);
  const float rstd = rsqrtf(var * (1.0f / 256.0f) + 1e-5f);
#pragma unroll
  for (int i = 0; i < 8; ++i) {
    const int d = i * 32 + lane;
    z[(size_t)wid * 256 + d] = f2bfu((v[i] - mu) * rstd * gam[d] + bet[d]);
  }
}

// ---------------------------------------------------------------------------
// Transposed f32 -> bf16 weight conversion: Wt[l][n][k] = bf16(W[l][k][n]).
// Makes WMMA B-operand K-pairs contiguous so GEMM staging is pure row copies.
// ---------------------------------------------------------------------------
__global__ __launch_bounds__(256) void cvtT_kernel(
    const float* __restrict__ W, unsigned short* __restrict__ Wt,
    int Kd, int Nd) {
  const int l = blockIdx.y;
  const int i = blockIdx.x * 256 + threadIdx.x;
  if (i >= Kd * Nd) return;
  const int k = i / Nd, n = i % Nd;
  const size_t base = (size_t)l * Kd * Nd;
  Wt[base + (size_t)n * Kd + k] = f2bfu(W[base + i]);
}

// ---------------------------------------------------------------------------
// Tiled bf16 WMMA GEMM: C[M,N] = A[M,K] @ B[K,N] + bias, fused epilogue.
// Block = 4 waves, tile 64x64, K-step 32.  A row-major [M][K]; Bt is the
// pre-transposed weight [N][K].  A tile: per-lane async global->LDS copies.
// B tile: one TDM tensor_load_to_lds per K-step (wave 0), LDS rows padded to
// 48 elems (96 B) by the TDM pad unit so 16 B accesses stay aligned.
// EPI: 0 = bf16 out, 1 = bf16(x^2), 2 = bf16(erf-gelu(x)), 3 = f32 outF += x
// ---------------------------------------------------------------------------
template <int EPI>
__global__ __launch_bounds__(128) void gemm_bf16_kernel(
    const unsigned short* __restrict__ A, const unsigned short* __restrict__ Bt,
    const float* __restrict__ bias, float* __restrict__ outF,
    unsigned short* __restrict__ outB, int M, int N, int K) {
  __shared__ unsigned short As[64][48];   // [m][k], 96 B rows (16 B aligned)
  __shared__ unsigned short Bs[64][48];   // [n][k]

  const int tid  = threadIdx.x;
  const int lane = tid & 31, w = tid >> 5;
  const int lr   = lane & 15, half = lane >> 4;
  const int m0   = blockIdx.x * 64, n0 = blockIdx.y * 64;
  const int r    = tid >> 1, hk = (tid & 1) * 16;   // staging row / k-half

  v8f acc0 = {}, acc1 = {}, acc2 = {}, acc3 = {};

  for (int kt = 0; kt < K; kt += 32) {
    // B tile 64x32 via Tensor Data Mover (one descriptor, wave 0 issues)
    if (tid < 32) {
      tdm_load_64x32_bf16(&Bs[0][0], Bt + (size_t)n0 * K + kt, K);
    }
    // A tile 64x32 via async global->LDS (16 bf16 = 2 x b128 per thread)
    {
      const unsigned short* ga = A + (size_t)(m0 + r) * K + kt + hk;
      async_g2l_b128(&As[r][hk],     ga);
      async_g2l_b128(&As[r][hk + 8], ga + 8);
    }
    if (kt + 32 < K) {  // prefetch next tiles (global_prefetch_b8)
      __builtin_prefetch(A + (size_t)(m0 + r) * K + kt + 32, 0, 1);
      __builtin_prefetch(Bt + (size_t)(n0 + r) * K + kt + 32, 0, 1);
    }
    wait_async0();
    __builtin_amdgcn_s_wait_tensorcnt(0);
    __syncthreads();

    v16bf a = {};
#pragma unroll
    for (int v = 0; v < 8; ++v) {  // A-operand layout (16-bit, 16x32)
      const int ka = (v & 3) * 2 + 8 * half + 16 * (v >> 2);
      put2(a, 2 * v, *(const unsigned int*)&As[w * 16 + lr][ka]);
    }
#pragma unroll
    for (int di = 0; di < 4; ++di) {
      v16bf b = {};
#pragma unroll
      for (int v = 0; v < 8; ++v) {  // B-operand layout (16-bit, 32x16)
        const int kb = (v & 3) * 2 + 8 * half + 16 * (v >> 2);
        put2(b, 2 * v, *(const unsigned int*)&Bs[di * 16 + lr][kb]);
      }
      v8f& acc = di == 0 ? acc0 : di == 1 ? acc1 : di == 2 ? acc2 : acc3;
      acc = wmma_bf16(a, b, acc);
    }
    __syncthreads();
  }

#pragma unroll
  for (int di = 0; di < 4; ++di) {
    const v8f& acc = di == 0 ? acc0 : di == 1 ? acc1 : di == 2 ? acc2 : acc3;
    const int gc = n0 + di * 16 + lr;
    const float bsv = bias[gc];
#pragma unroll
    for (int j = 0; j < 8; ++j) {
      const int gr = m0 + w * 16 + j + 8 * half;   // C layout: M = j + 8*half
      const float xv = acc[j] + bsv;
      const size_t idx = (size_t)gr * N + gc;
      if (EPI == 0)      outB[idx] = f2bfu(xv);
      else if (EPI == 1) outB[idx] = f2bfu(xv * xv);
      else if (EPI == 2) outB[idx] =
          f2bfu(0.5f * xv * (1.0f + erff(xv * 0.70710678118654752f)));
      else               outF[idx] += xv;
    }
  }
}

// ---------------------------------------------------------------------------
// Chunked causal linear attention (Performer scan), chunk C = 32.
// One wave per (batch, head) chain; 32x32 f32 state S held in 4 WMMA
// accumulators.  Per chunk, all math is 16x16x32 bf16 WMMA:
//   A  = tril(Q2 @ K2^T)                (3 live quadrants)
//   num = Q2 @ S_prev + A @ V           (8 WMMA)
//   den = rowsum(A) + Q2 . ksum_prev
//   S  += K2^T @ V                      (4 WMMA), ksum += colsum(K2)
//   h  += num / (den + 1e-16)
// ---------------------------------------------------------------------------
__global__ __launch_bounds__(32) void attn_scan_kernel(
    const unsigned short* __restrict__ q2, const unsigned short* __restrict__ k2,
    const unsigned short* __restrict__ vv, float* h) {
  const int chain = blockIdx.x;      // 0..31 = b*8 + head
  const int b     = chain >> 3;
  const int head  = chain & 7;
  const int lane  = threadIdx.x;
  const int lr    = lane & 15;
  const int half  = lane >> 4;

  __shared__ unsigned short Vt[32][33];    // V^T  [d][token]
  __shared__ unsigned short K2t[32][33];   // K2^T [d][token]
  __shared__ unsigned short Am[32][33];    // masked attn [q][k]
  __shared__ unsigned short Sbt[32][33];   // S^T  [d][m]
  __shared__ float ssum_l[32];
  __shared__ float den_l[32];

  v8f S00 = {}, S01 = {}, S10 = {}, S11 = {};
  float ssum = 0.0f;

  const size_t base = (size_t)b * 2048 * 256 + head * 32;

  for (int c = 0; c < 64; ++c) {
    const int t0 = c * 32;

    // stage V^T and K2^T for this chunk (lane = token)
    {
      const size_t r = base + (size_t)(t0 + lane) * 256;
#pragma unroll
      for (int j = 0; j < 16; ++j) {
        const unsigned int pv = *(const unsigned int*)(vv + r + 2 * j);
        Vt[2 * j][lane]     = (unsigned short)(pv & 0xffffu);
        Vt[2 * j + 1][lane] = (unsigned short)(pv >> 16);
        const unsigned int pk = *(const unsigned int*)(k2 + r + 2 * j);
        K2t[2 * j][lane]     = (unsigned short)(pk & 0xffffu);
        K2t[2 * j + 1][lane] = (unsigned short)(pk >> 16);
      }
    }
    // stage prefix state S^T as bf16, publish prefix ksum
#pragma unroll
    for (int j = 0; j < 8; ++j) {
      const int m = j + 8 * half;
      Sbt[lr][m]           = f2bfu(S00[j]);
      Sbt[16 + lr][m]      = f2bfu(S01[j]);
      Sbt[lr][16 + m]      = f2bfu(S10[j]);
      Sbt[16 + lr][16 + m] = f2bfu(S11[j]);
    }
    ssum_l[lane] = ssum;
    __syncthreads();

    // load Q2 (A-operand) and K2 (B-operand) straight from global
    v16bf aQ0 = {}, aQ1 = {}, bK0 = {}, bK1 = {};
#pragma unroll
    for (int v = 0; v < 8; ++v) {
      const int ka = (v & 3) * 2 + 8 * half + 16 * (v >> 2);
      put2(aQ0, 2 * v, *(const unsigned int*)(q2 + base + (size_t)(t0 + lr) * 256 + ka));
      put2(aQ1, 2 * v, *(const unsigned int*)(q2 + base + (size_t)(t0 + 16 + lr) * 256 + ka));
      const int kb = (v & 3) * 2 + 8 * half + 16 * (v >> 2);
      put2(bK0, 2 * v, *(const unsigned int*)(k2 + base + (size_t)(t0 + lr) * 256 + kb));
      put2(bK1, 2 * v, *(const unsigned int*)(k2 + base + (size_t)(t0 + 16 + lr) * 256 + kb));
    }

    // A = Q2 @ K2^T  (quadrant (0,1) is fully masked -> skipped)
    v8f A00 = {}, A10 = {}, A11 = {};
    A00 = wmma_bf16(aQ0, bK0, A00);
    A10 = wmma_bf16(aQ1, bK0, A10);
    A11 = wmma_bf16(aQ1, bK1, A11);

    // causal mask (keep key <= query) on diagonal quadrants
#pragma unroll
    for (int j = 0; j < 8; ++j) {
      const int M = j + 8 * half;
      if (lr > M) { A00[j] = 0.0f; A11[j] = 0.0f; }
    }

    // den intra part: row sums of masked A (16-lane shuffle reductions)
#pragma unroll
    for (int j = 0; j < 8; ++j) {
      const int M = j + 8 * half;
      float r0 = A00[j];
      float r1 = A10[j] + A11[j];
#pragma unroll
      for (int m = 1; m < 16; m <<= 1) {
        r0 += __shfl_xor(r0, m);
        r1 += __shfl_xor(r1, m);
      }
      if (lr == 0) { den_l[M] = r0; den_l[16 + M] = r1; }
    }

    // stage masked A as bf16 (zero the (0,1) quadrant)
#pragma unroll
    for (int j = 0; j < 8; ++j) {
      const int M = j + 8 * half;
      Am[M][lr]           = f2bfu(A00[j]);
      Am[M][16 + lr]      = 0;
      Am[16 + M][lr]      = f2bfu(A10[j]);
      Am[16 + M][16 + lr] = f2bfu(A11[j]);
    }
    __syncthreads();

    // den inter part: q2[t] . ksum_prefix   (lane = token)
    {
      float dv = den_l[lane];
      const unsigned short* qrow = q2 + base + (size_t)(t0 + lane) * 256;
#pragma unroll 8
      for (int m = 0; m < 32; ++m) dv += bf2f(qrow[m]) * ssum_l[m];
      den_l[lane] = dv + 1e-16f;
    }
    __syncthreads();

    // assemble LDS operands: S^T, V^T (B-operands), masked A (A-operand)
    v16bf bS0 = {}, bS1 = {}, bV0 = {}, bV1 = {}, aA0 = {}, aA1 = {};
#pragma unroll
    for (int v = 0; v < 8; ++v) {
      const int kb = (v & 3) * 2 + 8 * half + 16 * (v >> 2);
      bS0[2 * v] = usbf(Sbt[lr][kb]);        bS0[2 * v + 1] = usbf(Sbt[lr][kb + 1]);
      bS1[2 * v] = usbf(Sbt[16 + lr][kb]);   bS1[2 * v + 1] = usbf(Sbt[16 + lr][kb + 1]);
      bV0[2 * v] = usbf(Vt[lr][kb]);         bV0[2 * v + 1] = usbf(Vt[lr][kb + 1]);
      bV1[2 * v] = usbf(Vt[16 + lr][kb]);    bV1[2 * v + 1] = usbf(Vt[16 + lr][kb + 1]);
      const int ka = (v & 3) * 2 + 8 * half + 16 * (v >> 2);
      aA0[2 * v] = usbf(Am[lr][ka]);         aA0[2 * v + 1] = usbf(Am[lr][ka + 1]);
      aA1[2 * v] = usbf(Am[16 + lr][ka]);    aA1[2 * v + 1] = usbf(Am[16 + lr][ka + 1]);
    }

    // num = Q2 @ S_prev + tril(A) @ V
    v8f N00 = {}, N01 = {}, N10 = {}, N11 = {};
    N00 = wmma_bf16(aQ0, bS0, N00); N00 = wmma_bf16(aA0, bV0, N00);
    N01 = wmma_bf16(aQ0, bS1, N01); N01 = wmma_bf16(aA0, bV1, N01);
    N10 = wmma_bf16(aQ1, bS0, N10); N10 = wmma_bf16(aA1, bV0, N10);
    N11 = wmma_bf16(aQ1, bS1, N11); N11 = wmma_bf16(aA1, bV1, N11);

    // h += num / den
#pragma unroll
    for (int j = 0; j < 8; ++j) {
      const int M = j + 8 * half;
      const float d0 = den_l[M];
      const float d1 = den_l[16 + M];
      float* p0 = h + base + (size_t)(t0 + M) * 256;
      float* p1 = h + base + (size_t)(t0 + 16 + M) * 256;
      p0[lr]      += N00[j] / d0;
      p0[16 + lr] += N01[j] / d0;
      p1[lr]      += N10[j] / d1;
      p1[16 + lr] += N11[j] / d1;
    }

    // S += K2^T @ V
    v16bf aK0 = {}, aK1 = {};
#pragma unroll
    for (int v = 0; v < 8; ++v) {
      const int ka = (v & 3) * 2 + 8 * half + 16 * (v >> 2);
      aK0[2 * v] = usbf(K2t[lr][ka]);        aK0[2 * v + 1] = usbf(K2t[lr][ka + 1]);
      aK1[2 * v] = usbf(K2t[16 + lr][ka]);   aK1[2 * v + 1] = usbf(K2t[16 + lr][ka + 1]);
    }
    S00 = wmma_bf16(aK0, bV0, S00);
    S01 = wmma_bf16(aK0, bV1, S01);
    S10 = wmma_bf16(aK1, bV0, S10);
    S11 = wmma_bf16(aK1, bV1, S11);

    // ksum += column sums of K2 chunk (lane = feature m)
    {
      float s = 0.0f;
#pragma unroll 8
      for (int t = 0; t < 32; ++t) s += bf2f(K2t[lane][t]);
      ssum += s;
    }
    __syncthreads();
  }
}

// ---------------------------------------------------------------------------
// Output head: out[t] = h[t] . Wo + bo   (wave per token)
// ---------------------------------------------------------------------------
__global__ __launch_bounds__(256) void head_kernel(
    const float* __restrict__ h, const float* __restrict__ Wo,
    const float* __restrict__ bo, float* __restrict__ out, int T) {
  const int wid  = (blockIdx.x * blockDim.x + threadIdx.x) >> 5;
  const int lane = threadIdx.x & 31;
  if (wid >= T) return;
  const float* row = h + (size_t)wid * 256;
  float s = 0.0f;
#pragma unroll
  for (int i = 0; i < 8; ++i) s += row[i * 32 + lane] * Wo[i * 32 + lane];
#pragma unroll
  for (int m = 16; m >= 1; m >>= 1) s += __shfl_xor(s, m);
  if (lane == 0) out[wid] = s + bo[0];
}

// ---------------------------------------------------------------------------
extern "C" void kernel_launch(void* const* d_in, const int* in_sizes, int n_in,
                              void* d_out, int out_size, void* d_ws, size_t ws_size,
                              hipStream_t stream) {
  (void)in_sizes; (void)n_in; (void)out_size; (void)ws_size;

  const float* x    = (const float*)d_in[0];
  const float* ge   = (const float*)d_in[1];
  const float* Wq   = (const float*)d_in[2];
  const float* bq   = (const float*)d_in[3];
  const float* Wk   = (const float*)d_in[4];
  const float* bk   = (const float*)d_in[5];
  const float* Wv   = (const float*)d_in[6];
  const float* bv   = (const float*)d_in[7];
  const float* ln1g = (const float*)d_in[8];
  const float* ln1b = (const float*)d_in[9];
  const float* ln2g = (const float*)d_in[10];
  const float* ln2b = (const float*)d_in[11];
  const float* WU   = (const float*)d_in[12];
  const float* bU   = (const float*)d_in[13];
  const float* WVw  = (const float*)d_in[14];
  const float* bVd  = (const float*)d_in[15];
  const float* Wo   = (const float*)d_in[16];
  const float* bo   = (const float*)d_in[17];
  float* out = (float*)d_out;

  constexpr int T = 8192, D = 256, F = 1024, L = 4;

  char* p = (char*)d_ws;
  float* h            = (float*)p;          p += (size_t)T * D * 4;
  unsigned short* z   = (unsigned short*)p; p += (size_t)T * D * 2;
  unsigned short* q2  = (unsigned short*)p; p += (size_t)T * D * 2;
  unsigned short* k2b = (unsigned short*)p; p += (size_t)T * D * 2;
  unsigned short* vb  = (unsigned short*)p; p += (size_t)T * D * 2;
  unsigned short* ub  = (unsigned short*)p; p += (size_t)T * F * 2;
  unsigned short* tWq = (unsigned short*)p; p += (size_t)L * D * D * 2;  // [n][k]
  unsigned short* tWk = (unsigned short*)p; p += (size_t)L * D * D * 2;
  unsigned short* tWv = (unsigned short*)p; p += (size_t)L * D * D * 2;
  unsigned short* tWU = (unsigned short*)p; p += (size_t)L * D * F * 2;  // [F][D]
  unsigned short* tWV = (unsigned short*)p; p += (size_t)L * F * D * 2;  // [D][F]

  // transposed bf16 weight prep (deterministic: done every launch)
  {
    dim3 gdd((D * D + 255) / 256, L);
    dim3 gdf((D * F + 255) / 256, L);
    cvtT_kernel<<<gdd, 256, 0, stream>>>(Wq, tWq, D, D);
    cvtT_kernel<<<gdd, 256, 0, stream>>>(Wk, tWk, D, D);
    cvtT_kernel<<<gdd, 256, 0, stream>>>(Wv, tWv, D, D);
    cvtT_kernel<<<gdf, 256, 0, stream>>>(WU, tWU, D, F);   // -> [F][D]
    cvtT_kernel<<<gdf, 256, 0, stream>>>(WVw, tWV, F, D);  // -> [D][F]
  }

  embed_kernel<<<T, D, 0, stream>>>(x, ge, h);

  const dim3 gqkv(T / 64, D / 64);
  const dim3 gup(T / 64, F / 64);
  const dim3 gdn(T / 64, D / 64);

  for (int l = 0; l < L; ++l) {
    ln_kernel<<<T / 8, 256, 0, stream>>>(h, ln1g + l * D, ln1b + l * D, z, T);
    gemm_bf16_kernel<1><<<gqkv, 128, 0, stream>>>(z, tWq + (size_t)l * D * D,
        bq + l * D, nullptr, q2, T, D, D);
    gemm_bf16_kernel<1><<<gqkv, 128, 0, stream>>>(z, tWk + (size_t)l * D * D,
        bk + l * D, nullptr, k2b, T, D, D);
    gemm_bf16_kernel<0><<<gqkv, 128, 0, stream>>>(z, tWv + (size_t)l * D * D,
        bv + l * D, nullptr, vb, T, D, D);
    attn_scan_kernel<<<32, 32, 0, stream>>>(q2, k2b, vb, h);
    ln_kernel<<<T / 8, 256, 0, stream>>>(h, ln2g + l * D, ln2b + l * D, z, T);
    gemm_bf16_kernel<2><<<gup, 128, 0, stream>>>(z, tWU + (size_t)l * D * F,
        bU + l * F, nullptr, ub, T, F, D);
    gemm_bf16_kernel<3><<<gdn, 128, 0, stream>>>(ub, tWV + (size_t)l * F * D,
        bVd + l * D, h, nullptr, T, D, F);
  }

  head_kernel<<<T / 8, 256, 0, stream>>>(h, Wo, bo, out, T);
}
